// MHA_7842610283168
// MI455X (gfx1250) — matrix operand order repaired and checked by reading, not compiled, not run on hardware
//
#include <hip/hip_runtime.h>
#include <hip/hip_bf16.h>

typedef __attribute__((ext_vector_type(16))) _Float16     v16h;
typedef __attribute__((ext_vector_type(8)))  _Float16     v8h;
typedef __attribute__((ext_vector_type(8)))  float        v8f;
typedef __attribute__((ext_vector_type(4)))  unsigned int v4u;
typedef __attribute__((ext_vector_type(8)))  int          v8i;
typedef __attribute__((ext_vector_type(4)))  int          v4i;

#define DIMM 1024
#define SEQ  2048
#define NB   2
#define NH   16
#define HD   64

// ---------------------------------------------------------------------------
// fp32 -> f16 convert
// ---------------------------------------------------------------------------
__global__ void mha_cvt_f16(const float* __restrict__ src,
                            _Float16* __restrict__ dst, int n) {
    int i = blockIdx.x * blockDim.x + threadIdx.x;
    if (i < n) dst[i] = (_Float16)src[i];
}

// ---------------------------------------------------------------------------
// RoPE (reference: x1=x[0::2], x2=x[1::2]; out = [x1*c - x2*s, x1*s + x2*c])
// ---------------------------------------------------------------------------
__global__ void mha_rope(const _Float16* __restrict__ src,
                         const float* __restrict__ theta,
                         _Float16* __restrict__ dst, int total) {
    int i = blockIdx.x * blockDim.x + threadIdx.x;
    if (i >= total) return;
    int j   = i & 31;
    int tmp = i >> 5;
    int h   = tmp & 15;
    tmp >>= 4;
    int s = tmp & (SEQ - 1);
    int b = tmp >> 11;
    float ang = (float)s * theta[j];
    float c = cosf(ang), sn = sinf(ang);
    size_t base = ((size_t)(b * SEQ + s)) * DIMM + h * HD;
    float x1 = (float)src[base + 2 * j];
    float x2 = (float)src[base + 2 * j + 1];
    dst[base + j]      = (_Float16)(x1 * c  - x2 * sn);
    dst[base + j + 32] = (_Float16)(x1 * sn + x2 * c);
}

// ---------------------------------------------------------------------------
// A-fragment loader: 16-bit A 16x32 layout (lane m=l&15, hi=l>>4):
//   halves 0..7  from p[0..7]   (K = kb + 8*hi + e)
//   halves 8..15 from p[16..23] (K = kb + 16 + 8*hi + e)
// ---------------------------------------------------------------------------
__device__ __forceinline__ v16h mha_load_a(const _Float16* p) {
    v8h lo = *(const v8h*)p;
    v8h hv = *(const v8h*)(p + 16);
    v16h r;
    #pragma unroll
    for (int i = 0; i < 8; ++i) { r[i] = lo[i]; r[i + 8] = hv[i]; }
    return r;
}

// ---------------------------------------------------------------------------
// WMMA GEMM: out[M,N] = A[M,K] @ W[K,N] + bias.   Requires K % 64 == 0.
// Each wave: 32(M) x 64(N) tile (8 accumulators); B-fragments reused across
// both 16-row halves. K-loop manually unrolled x2 with compile-time ping-pong
// fragment sets, so next-step global_load_b128s land directly in the alternate
// registers (no rotation moves) and overlap the current WMMA chain.
// ---------------------------------------------------------------------------
__global__ __launch_bounds__(128) void mha_gemm_wmma(
    const _Float16* __restrict__ A, const _Float16* __restrict__ W,
    const float* __restrict__ bias,
    _Float16* __restrict__ outh, float* __restrict__ outf,
    int M, int N, int K, int write_f32)
{
    const int lane = threadIdx.x & 31;
    const int wave = threadIdx.x >> 5;
    const int m0   = blockIdx.x * 32;
    const int n0   = blockIdx.y * 256 + wave * 64;
    const int mrow = lane & 15;
    const int hi   = lane >> 4;

    v8f acc[2][4];
    #pragma unroll
    for (int h2 = 0; h2 < 2; ++h2)
        #pragma unroll
        for (int t = 0; t < 4; ++t)
            #pragma unroll
            for (int i = 0; i < 8; ++i) acc[h2][t][i] = 0.0f;

    const _Float16* ap0 = A + (size_t)(m0 + mrow) * K + 8 * hi;
    const _Float16* ap1 = ap0 + (size_t)16 * K;
    const _Float16* wp  = W + (size_t)(8 * hi) * N + n0 + mrow;

    // ping-pong fragment sets
    v16h a0[2], a1[2], bf[2][4];

    // prologue: set 0 holds kb = 0
    a0[0] = mha_load_a(ap0);
    a1[0] = mha_load_a(ap1);
    #pragma unroll
    for (int t = 0; t < 4; ++t)
        for (int e = 0; e < 16; ++e) bf[0][t][e] = wp[(size_t)(e + (e & 8)) * N + 16 * t];

    for (int kb = 0; kb < K; kb += 64) {
        #pragma unroll
        for (int ping = 0; ping < 2; ++ping) {
            const int pong = ping ^ 1;
            const int knxt = kb + 32 * ping + 32;
            if (knxt < K) {
                a0[pong] = mha_load_a(ap0 + knxt);
                a1[pong] = mha_load_a(ap1 + knxt);
                const _Float16* wpn = wp + (size_t)knxt * N;
                __builtin_prefetch(wpn + (size_t)32 * N, 0, 1); // global_prefetch_b8
                #pragma unroll
                for (int t = 0; t < 4; ++t)
                    for (int e = 0; e < 16; ++e)
                        bf[pong][t][e] = wpn[(size_t)(e + (e & 8)) * N + 16 * t];
            }
            #pragma unroll
            for (int t = 0; t < 4; ++t) {
                acc[0][t] = __builtin_amdgcn_wmma_f32_16x16x32_f16(
                    false, a0[ping], false, bf[ping][t], (short)0,
                    acc[0][t], false, false);
                acc[1][t] = __builtin_amdgcn_wmma_f32_16x16x32_f16(
                    false, a1[ping], false, bf[ping][t], (short)0,
                    acc[1][t], false, false);
            }
        }
    }

    #pragma unroll
    for (int h2 = 0; h2 < 2; ++h2)
        #pragma unroll
        for (int t = 0; t < 4; ++t) {
            int col = n0 + 16 * t + mrow;
            float bv = bias[col];
            #pragma unroll
            for (int v = 0; v < 8; ++v) {
                int row = m0 + 16 * h2 + v + 8 * hi;
                float val = acc[h2][t][v] + bv;
                if (write_f32) outf[(size_t)row * N + col] = val;
                else           outh[(size_t)row * N + col] = (_Float16)val;
            }
        }
}

// ---------------------------------------------------------------------------
// TDM stage: 32 keys x 64 dims of f16 (row stride DIMM elems) -> LDS, packed
// row-major. One tensor_load_to_lds per wave; descriptors per ISA 08 §8.3/8.4.
// ---------------------------------------------------------------------------
__device__ __forceinline__ void mha_tdm_load_k(unsigned lds_addr,
                                               const _Float16* gptr) {
    unsigned long long ga = (unsigned long long)(size_t)gptr;
    v4u g0;
    g0[0] = 1u;                                   // count=1, user descriptor
    g0[1] = lds_addr;                             // lds_addr (bytes)
    g0[2] = (unsigned)(ga & 0xFFFFFFFFu);         // global_addr[31:0]
    g0[3] = (unsigned)((ga >> 32) & 0x01FFFFFFu)  // global_addr[56:32]
          | (2u << 30);                           // type = 2 ("image")
    v8i g1;
    g1[0] = 0x00010000;        // workgroup_mask=0, data_size=1 (2 bytes)
    g1[1] = (int)(64u << 16);  // tensor_dim0[15:0] = 64
    g1[2] = (int)(32u << 16);  // tensor_dim0 hi=0 | tensor_dim1[15:0] = 32
    g1[3] = (int)(64u << 16);  // tensor_dim1 hi=0 | tile_dim0 = 64
    g1[4] = 32;                // tile_dim1 = 32, tile_dim2 = 0
    g1[5] = (int)DIMM;         // tensor_dim0_stride[31:0] = 1024
    g1[6] = 0;                 // stride hi, tensor_dim1_stride lo
    g1[7] = 0;
    v4i g2, g3;
    #pragma unroll
    for (int i = 0; i < 4; ++i) { g2[i] = 0; g3[i] = 0; }
#if __clang_major__ >= 23
    v8i g4;
    #pragma unroll
    for (int i = 0; i < 8; ++i) g4[i] = 0;
    __builtin_amdgcn_tensor_load_to_lds(g0, g1, g2, g3, g4, 0);
#else
    __builtin_amdgcn_tensor_load_to_lds(g0, g1, g2, g3, 0);
#endif
}

// ---------------------------------------------------------------------------
// Flash attention, causal. One wave32 per (b, h, 16-row q tile).
// K tile staged by the Tensor Data Mover (tensor_load_to_lds + tensorcnt).
// ---------------------------------------------------------------------------
__global__ __launch_bounds__(32) void mha_flash_attn(
    const _Float16* __restrict__ Q, const _Float16* __restrict__ Kr,
    const _Float16* __restrict__ V, _Float16* __restrict__ ctx)
{
    __shared__ __align__(32) _Float16 lds_k[32 * HD];   // 32 keys x 64 dims
    __shared__ __align__(32) _Float16 lds_p[16 * 32];   // 16 q-rows x 32 keys

    const int lane = threadIdx.x;
    const int bid  = blockIdx.x;
    const int qt   = bid & 127;
    const int h    = (bid >> 7) & 15;
    const int b    = bid >> 11;
    const int q0   = qt * 16;
    const int mrow = lane & 15;
    const int hi   = lane >> 4;

    // LDS aperture: low 32 bits of the generic pointer are the LDS byte addr
    const unsigned lds_k_addr = (unsigned)(size_t)(&lds_k[0]);

    // Q fragments for both 32-wide K-steps of head_dim
    v16h qa[2];
    #pragma unroll
    for (int ks = 0; ks < 2; ++ks)
        qa[ks] = mha_load_a(Q + ((size_t)(b * SEQ + q0 + mrow)) * DIMM
                              + h * HD + ks * 32 + 8 * hi);

    v8f acc[4];
    float rmax[8], rsum[8];
    #pragma unroll
    for (int t = 0; t < 4; ++t)
        #pragma unroll
        for (int i = 0; i < 8; ++i) acc[t][i] = 0.0f;
    #pragma unroll
    for (int v = 0; v < 8; ++v) { rmax[v] = -1e30f; rsum[v] = 0.0f; }

    for (int kb2 = 0; kb2 <= q0 + 15; kb2 += 32) {
        // ---- TDM: DMA the 32x64 K block into LDS, wait on TENSORcnt ----
        mha_tdm_load_k(lds_k_addr,
                       Kr + ((size_t)(b * SEQ + kb2)) * DIMM + h * HD);
        __builtin_amdgcn_s_wait_tensorcnt(0);
        __syncthreads();

        float p[2][8];
        float bm[8];
        #pragma unroll
        for (int v = 0; v < 8; ++v) bm[v] = -1e30f;

        const int kd = mrow + 16 * hi;  // this lane's K-row inside a 32-step
        #pragma unroll
        for (int t = 0; t < 2; ++t) {
            v8f s;
            #pragma unroll
            for (int i = 0; i < 8; ++i) s[i] = 0.0f;
            #pragma unroll
            for (int ks = 0; ks < 2; ++ks) {
                v16h kf;  // B = K^T: lane holds key 16t+mrow, elems over head-dims
                #pragma unroll
                for (int e = 0; e < 16; ++e)
                    kf[e] = lds_k[(16 * t + mrow) * HD + ks * 32 + 8 * hi + e + (e & 8)];
                s = __builtin_amdgcn_wmma_f32_16x16x32_f16(
                    false, qa[ks], false, kf, (short)0, s, false, false);
            }
            const int kj = kb2 + 16 * t + mrow;
            #pragma unroll
            for (int v = 0; v < 8; ++v) {
                int qi = q0 + v + 8 * hi;
                float sv = s[v] * 0.125f;          // 1/sqrt(64)
                if (kj > qi) sv = -1e30f;          // causal mask
                p[t][v] = sv;
                bm[v] = fmaxf(bm[v], sv);
            }
        }

        // row max over the 16 lanes holding this row (wave32-safe masks)
        #pragma unroll
        for (int v = 0; v < 8; ++v) {
            float x = bm[v];
            x = fmaxf(x, __shfl_xor(x, 1));
            x = fmaxf(x, __shfl_xor(x, 2));
            x = fmaxf(x, __shfl_xor(x, 4));
            x = fmaxf(x, __shfl_xor(x, 8));
            bm[v] = x;
        }

        #pragma unroll
        for (int v = 0; v < 8; ++v) {
            float mnew  = fmaxf(rmax[v], bm[v]);
            float scale = __expf(rmax[v] - mnew);
            float e0 = __expf(p[0][v] - mnew);
            float e1 = __expf(p[1][v] - mnew);
            p[0][v] = e0; p[1][v] = e1;
            float sum = e0 + e1;
            sum += __shfl_xor(sum, 1);
            sum += __shfl_xor(sum, 2);
            sum += __shfl_xor(sum, 4);
            sum += __shfl_xor(sum, 8);
            rsum[v] = rsum[v] * scale + sum;
            rmax[v] = mnew;
            #pragma unroll
            for (int t = 0; t < 4; ++t) acc[t][v] *= scale;
        }

        // P: C-layout -> LDS row-major
        #pragma unroll
        for (int t = 0; t < 2; ++t)
            #pragma unroll
            for (int v = 0; v < 8; ++v)
                lds_p[(v + 8 * hi) * 32 + 16 * t + mrow] = (_Float16)p[t][v];
        __syncthreads();

        // P A-fragment (M=16 q rows, K=32 keys)
        v16h pa = mha_load_a(lds_p + mrow * 32 + 8 * hi);

        // V B-fragments: lane holds dim column 16t+mrow, elems over keys
        const _Float16* vp = V + ((size_t)(b * SEQ + kb2 + 8 * hi)) * DIMM + h * HD + mrow;
        #pragma unroll
        for (int t = 0; t < 4; ++t) {
            v16h vf;
            for (int e = 0; e < 16; ++e) vf[e] = vp[(size_t)(e + (e & 8)) * DIMM + 16 * t];
            acc[t] = __builtin_amdgcn_wmma_f32_16x16x32_f16(
                false, pa, false, vf, (short)0, acc[t], false, false);
        }
        __syncthreads();
    }

    // normalize and write context [b, s, h*64 + d] as f16
    #pragma unroll
    for (int t = 0; t < 4; ++t)
        #pragma unroll
        for (int v = 0; v < 8; ++v) {
            float o = acc[t][v] / rsum[v];
            ctx[((size_t)(b * SEQ + q0 + v + 8 * hi)) * DIMM
                + h * HD + 16 * t + mrow] = (_Float16)o;
        }
}

// ---------------------------------------------------------------------------
// host-side launcher
// ---------------------------------------------------------------------------
extern "C" void kernel_launch(void* const* d_in, const int* in_sizes, int n_in,
                              void* d_out, int out_size, void* d_ws, size_t ws_size,
                              hipStream_t stream) {
    (void)in_sizes; (void)n_in; (void)out_size; (void)ws_size;

    const float* x     = (const float*)d_in[0];
    /* positions d_in[1] unused: reference internally uses arange(seq_len) */
    const float* theta = (const float*)d_in[2];
    const float* Wq    = (const float*)d_in[3];
    const float* bq    = (const float*)d_in[4];
    const float* Wk    = (const float*)d_in[5];
    const float* bk    = (const float*)d_in[6];
    const float* Wv    = (const float*)d_in[7];
    const float* bv    = (const float*)d_in[8];
    const float* Wo    = (const float*)d_in[9];
    const float* bo    = (const float*)d_in[10];
    float* out = (float*)d_out;

    const int M  = NB * SEQ;           // 4096
    const int XN = M * DIMM;           // 4,194,304 elems
    const int WN = DIMM * DIMM;        // 1,048,576 elems

    _Float16* ws = (_Float16*)d_ws;
    _Float16* xh   = ws;
    _Float16* wqh  = xh   + (size_t)XN;
    _Float16* wkh  = wqh  + (size_t)WN;
    _Float16* wvh  = wkh  + (size_t)WN;
    _Float16* woh  = wvh  + (size_t)WN;
    _Float16* qraw = woh  + (size_t)WN;
    _Float16* kraw = qraw + (size_t)XN;
    _Float16* vh   = kraw + (size_t)XN;
    _Float16* qr   = vh   + (size_t)XN;
    _Float16* kr   = qr   + (size_t)XN;
    _Float16* ctx  = kr   + (size_t)XN;   // total = 64 MB of f16

    // 1) convert inputs to f16
    mha_cvt_f16<<<XN / 256, 256, 0, stream>>>(x,  xh,  XN);
    mha_cvt_f16<<<WN / 256, 256, 0, stream>>>(Wq, wqh, WN);
    mha_cvt_f16<<<WN / 256, 256, 0, stream>>>(Wk, wkh, WN);
    mha_cvt_f16<<<WN / 256, 256, 0, stream>>>(Wv, wvh, WN);
    mha_cvt_f16<<<WN / 256, 256, 0, stream>>>(Wo, woh, WN);

    // 2) QKV projections (WMMA GEMM, f16 output); wave tile 32x64
    dim3 ggrid(M / 32, 4);
    mha_gemm_wmma<<<ggrid, 128, 0, stream>>>(xh, wqh, bq, qraw, nullptr,
                                             M, DIMM, DIMM, 0);
    mha_gemm_wmma<<<ggrid, 128, 0, stream>>>(xh, wkh, bk, kraw, nullptr,
                                             M, DIMM, DIMM, 0);
    mha_gemm_wmma<<<ggrid, 128, 0, stream>>>(xh, wvh, bv, vh, nullptr,
                                             M, DIMM, DIMM, 0);

    // 3) RoPE on Q and K (out-of-place)
    const int rtot = NB * SEQ * NH * (HD / 2);   // 2,097,152
    mha_rope<<<rtot / 256, 256, 0, stream>>>(qraw, theta, qr, rtot);
    mha_rope<<<rtot / 256, 256, 0, stream>>>(kraw, theta, kr, rtot);

    // 4) causal flash attention: one wave32 per (b,h,q-tile16), TDM-staged K
    mha_flash_attn<<<NB * NH * (SEQ / 16), 32, 0, stream>>>(qr, kr, vh, ctx);

    // 5) output projection (WMMA GEMM, f32 output + bias)
    mha_gemm_wmma<<<ggrid, 128, 0, stream>>>(ctx, woh, bo, nullptr, out,
                                             M, DIMM, DIMM, 1);
}
